// BiLISTM_CRF_58643483460267
// MI455X (gfx1250) — compile-verified, matrix operand order checked by reading, not run on hardware
//
#include <hip/hip_runtime.h>
#include <math.h>

typedef _Float16 half_t;
typedef __attribute__((ext_vector_type(16))) _Float16 v16h;
typedef __attribute__((ext_vector_type(8)))  float    v8f;

#define BATCH 256
#define TLEN  512
#define EMB   64
#define HID   128
#define G4    512   // 4*HID
#define TAGS  32

__device__ __forceinline__ float sigm(float x) { return 1.0f / (1.0f + __expf(-x)); }

__device__ __forceinline__ v8f wmma_f16(v16h a, v16h b, v8f c) {
    return __builtin_amdgcn_wmma_f32_16x16x32_f16(false, a, false, b, (short)0, c, false, false);
}

// 16x32 f16 fragment load (A layout; same striping reused for B via transposed
// weights). Row-major, leading dim ld. Lanes 0-15 = rows; lane-half picks the
// K sub-block -> two contiguous 16B loads per lane (ds/global _load_b128).
__device__ __forceinline__ v16h load_tile(const half_t* base, int ld, int lane) {
    int r  = lane & 15;
    int hi = (lane >> 4) & 1;
    const half_t* p = base + r * ld + hi * 8;
    v16h a;
#pragma unroll
    for (int e = 0; e < 8; ++e) { a[e] = p[e]; a[e + 8] = p[e + 16]; }
    return a;
}

// ---- prep: transpose Wk_{f,b} (64x512) -> f16 [512][64], Wr_{f,b} (128x512) -> f16 [512][128]
__global__ void prep_weights_kernel(const float* __restrict__ Wk_f, const float* __restrict__ Wk_b,
                                    const float* __restrict__ Wr_f, const float* __restrict__ Wr_b,
                                    half_t* __restrict__ WkT, half_t* __restrict__ WrT) {
    int i = blockIdx.x * 256 + threadIdx.x;
    if (i < 2 * G4 * EMB) {
        const float* Wk = (i < G4 * EMB) ? Wk_f : Wk_b;
        int ii = i & (G4 * EMB - 1);
        WkT[i] = (half_t)Wk[(ii & 63) * G4 + (ii >> 6)];
    }
    if (i < 2 * G4 * HID) {
        const float* Wr = (i < G4 * HID) ? Wr_f : Wr_b;
        int jj = i & (G4 * HID - 1);
        WrT[i] = (half_t)Wr[(jj & 127) * G4 + (jj >> 7)];
    }
}

// ---- bidirectional LSTM scan: 32 workgroups (16 fwd + 16 bwd), 8 waves each.
// Wave w owns gate-striped columns n0 = g*128 + w*16; ALL weight fragments
// (16 Wr + 8 Wk = 192 VGPRs) are hoisted into registers for the whole scan, so
// the 24 WMMAs per step wait only on the 12 A-fragment ds_load_b128s. Gate
// nonlinearity runs directly on accumulators. One barrier/step; Hs/Xs ping-pong.
__global__ __launch_bounds__(256)
void lstm_bidir_kernel(const int* __restrict__ inputs, const float* __restrict__ E,
                       const float* __restrict__ bias_f, const float* __restrict__ bias_b,
                       const half_t* __restrict__ WkT, const half_t* __restrict__ WrT,
                       half_t* __restrict__ h_cat) {
    __shared__ half_t Hs[2 * 16 * HID];   // ping-pong hidden state (f16)
    __shared__ half_t Xs[2 * 16 * EMB];   // ping-pong embeddings (f16)

    const int dir = blockIdx.x >> 4;             // 0 = forward, 1 = backward
    const int b0  = (blockIdx.x & 15) * 16;
    const float*  bias  = dir ? bias_b : bias_f;
    const half_t* WkT_g = WkT + (long)dir * G4 * EMB;
    const half_t* WrT_g = WrT + (long)dir * G4 * HID;

    const int tid  = threadIdx.x;
    const int lane = tid & 31;
    const int wv   = tid >> 5;          // wave 0..7
    const int c    = lane & 15;
    const int hi   = (lane >> 4) & 1;
    const int j    = wv * 16 + c;       // hidden unit owned by this lane
    const int r_g  = tid >> 4;          // gather row 0..15
    const int k4   = (tid & 15) * 4;    // gather col base

    // hoist ALL weight fragments into registers (loop-invariant for the scan)
    v16h wkf[4][2];
    v16h wrf[4][4];
#pragma unroll
    for (int g = 0; g < 4; ++g) {
        const int n0 = g * HID + wv * 16;
#pragma unroll
        for (int kc = 0; kc < 2; ++kc)
            wkf[g][kc] = load_tile(WkT_g + (long)n0 * EMB + kc * 32, EMB, lane);
#pragma unroll
        for (int kc = 0; kc < 4; ++kc)
            wrf[g][kc] = load_tile(WrT_g + (long)n0 * HID + kc * 32, HID, lane);
    }

    // per-lane biases (loop-invariant)
    const float bI = bias[j];
    const float bF = bias[HID + j];
    const float bG = bias[2 * HID + j];
    const float bO = bias[3 * HID + j];

    for (int i = tid; i < 16 * HID; i += 256) Hs[i] = (half_t)0.0f;

    float creg[8];
#pragma unroll
    for (int v = 0; v < 8; ++v) creg[v] = 0.0f;

    __syncthreads();

    for (int step = 0; step < TLEN; ++step) {
        const int t = dir ? (TLEN - 1 - step) : step;
        half_t* Xcur      = Xs + (step & 1) * (16 * EMB);
        const half_t* Hin = Hs + (step & 1) * (16 * HID);
        half_t* Hout      = Hs + ((step + 1) & 1) * (16 * HID);

        // gather embeddings for this step's 16 rows -> Xcur (f16)
        {
            int tok = inputs[(b0 + r_g) * TLEN + t];
            const float* er = E + (long)tok * EMB + k4;
            half_t* xr = Xcur + r_g * EMB + k4;
            xr[0] = (half_t)er[0]; xr[1] = (half_t)er[1];
            xr[2] = (half_t)er[2]; xr[3] = (half_t)er[3];
        }
        __syncthreads();   // single barrier per step

        // A fragments (shared across the 4 gate tiles)
        v16h xa0 = load_tile(Xcur,      EMB, lane);
        v16h xa1 = load_tile(Xcur + 32, EMB, lane);
        v16h ha0 = load_tile(Hin,       HID, lane);
        v16h ha1 = load_tile(Hin + 32,  HID, lane);
        v16h ha2 = load_tile(Hin + 64,  HID, lane);
        v16h ha3 = load_tile(Hin + 96,  HID, lane);

        v8f acc[4];
#pragma unroll
        for (int g = 0; g < 4; ++g) {
            v8f a = {0.f, 0.f, 0.f, 0.f, 0.f, 0.f, 0.f, 0.f};
            a = wmma_f16(xa0, wkf[g][0], a);
            a = wmma_f16(xa1, wkf[g][1], a);
            a = wmma_f16(ha0, wrf[g][0], a);
            a = wmma_f16(ha1, wrf[g][1], a);
            a = wmma_f16(ha2, wrf[g][2], a);
            a = wmma_f16(ha3, wrf[g][3], a);
            acc[g] = a;
        }

        // gate nonlinearity directly on accumulators; lane owns rows v+hi*8, col j
#pragma unroll
        for (int v = 0; v < 8; ++v) {
            float zi = acc[0][v] + bI;
            float zf = acc[1][v] + bF;
            float zg = acc[2][v] + bG;
            float zo = acc[3][v] + bO;
            float cc = sigm(zf) * creg[v] + sigm(zi) * tanhf(zg);
            float hh = sigm(zo) * tanhf(cc);
            creg[v] = cc;
            half_t hq = (half_t)hh;
            int row = v + hi * 8;
            Hout[row * HID + j] = hq;   // feeds next step's A operand
            h_cat[((long)(b0 + row) * TLEN + t) * (2 * HID) + dir * HID + j] = hq;
        }
    }
}

// ---------------- logits: (B*T,256) @ (256,32) + bd via WMMA ----------------
__global__ __launch_bounds__(64)
void logits_kernel(const half_t* __restrict__ h_cat, const float* __restrict__ Wd,
                   const float* __restrict__ bd, float* __restrict__ logits) {
    __shared__ half_t WdT[TAGS * 2 * HID];   // 32 x 256 f16 (transposed)
    __shared__ float  bds[TAGS];
    const int tid = threadIdx.x;
    for (int i = tid; i < TAGS * 2 * HID; i += 64) {
        int n = i >> 8, m = i & 255;
        WdT[i] = (half_t)Wd[m * TAGS + n];
    }
    if (tid < TAGS) bds[tid] = bd[tid];
    __syncthreads();

    const int lane = tid & 31;
    const int wv   = tid >> 5;          // wave 0/1 -> N-tile 0/1
    const long r0  = (long)blockIdx.x * 16;

    v8f acc = {0.f, 0.f, 0.f, 0.f, 0.f, 0.f, 0.f, 0.f};
#pragma unroll
    for (int kc = 0; kc < 8; ++kc) {
        v16h a = load_tile(h_cat + r0 * 256 + kc * 32, 256, lane);
        v16h b = load_tile(WdT + (wv * 16) * 256 + kc * 32, 256, lane);
        acc = wmma_f16(a, b, acc);
    }
    const int c = lane & 15, hi = (lane >> 4) & 1;
#pragma unroll
    for (int v = 0; v < 8; ++v)
        logits[(r0 + v + hi * 8) * TAGS + wv * 16 + c] = acc[v] + bds[wv * 16 + c];
}

// ---------------- CRF forward + score: one wave per batch row, lane = tag ----------------
__global__ __launch_bounds__(32)
void crf_kernel(const float* __restrict__ logits, const int* __restrict__ labels,
                const float* __restrict__ trans, float* __restrict__ out_ll) {
    __shared__ float Ts[TAGS * TAGS];
    __shared__ float sa[TAGS];
    __shared__ float red[TAGS];
    const int k = threadIdx.x;
    const int b = blockIdx.x;
    for (int i = k; i < TAGS * TAGS; i += TAGS) Ts[i] = trans[i];
    __syncthreads();

    const float* lg = logits + (long)b * TLEN * TAGS;
    float alpha = lg[k];
    for (int t = 1; t < TLEN; ++t) {
        sa[k] = alpha;
        __syncthreads();
        float m = -INFINITY;
#pragma unroll
        for (int jj = 0; jj < TAGS; ++jj) m = fmaxf(m, sa[jj] + Ts[jj * TAGS + k]);
        float s = 0.0f;
#pragma unroll
        for (int jj = 0; jj < TAGS; ++jj) s += __expf(sa[jj] + Ts[jj * TAGS + k] - m);
        alpha = m + __logf(s) + lg[t * TAGS + k];
        __syncthreads();
    }
    sa[k] = alpha;
    __syncthreads();
    float m = -INFINITY;
#pragma unroll
    for (int jj = 0; jj < TAGS; ++jj) m = fmaxf(m, sa[jj]);
    float s = 0.0f;
#pragma unroll
    for (int jj = 0; jj < TAGS; ++jj) s += __expf(sa[jj] - m);
    float log_norm = m + __logf(s);

    const int* lab = labels + b * TLEN;
    float acc = 0.0f;
    for (int t = k; t < TLEN; t += TAGS) acc += lg[t * TAGS + lab[t]];
    for (int t = k; t < TLEN - 1; t += TAGS) acc += Ts[lab[t] * TAGS + lab[t + 1]];
    red[k] = acc;
    __syncthreads();
    if (k == 0) {
        float tot = 0.0f;
        for (int jj = 0; jj < TAGS; ++jj) tot += red[jj];
        out_ll[b] = tot - log_norm;
    }
}

__global__ void copy_trans_kernel(const float* __restrict__ trans, float* __restrict__ out) {
    int i = blockIdx.x * 256 + threadIdx.x;
    if (i < TAGS * TAGS) out[BATCH + i] = trans[i];
}

extern "C" void kernel_launch(void* const* d_in, const int* in_sizes, int n_in,
                              void* d_out, int out_size, void* d_ws, size_t ws_size,
                              hipStream_t stream) {
    const int*   inputs = (const int*)d_in[0];
    const int*   labels = (const int*)d_in[1];
    const float* E      = (const float*)d_in[2];
    const float* Wk_f   = (const float*)d_in[3];
    const float* Wr_f   = (const float*)d_in[4];
    const float* b_f    = (const float*)d_in[5];
    const float* Wk_b   = (const float*)d_in[6];
    const float* Wr_b   = (const float*)d_in[7];
    const float* b_b    = (const float*)d_in[8];
    const float* Wd     = (const float*)d_in[9];
    const float* bd     = (const float*)d_in[10];
    const float* trans  = (const float*)d_in[11];
    float* out = (float*)d_out;

    // workspace carve
    char* ws = (char*)d_ws;
    half_t* h_cat  = (half_t*)ws;                                    // B*T*256 f16 = 64 MiB
    size_t  off    = (size_t)BATCH * TLEN * 2 * HID * sizeof(half_t);
    float*  logits = (float*)(ws + off);                             // B*T*32 f32 = 16 MiB
    off += (size_t)BATCH * TLEN * TAGS * sizeof(float);
    half_t* WkT    = (half_t*)(ws + off);                            // 2 x 512x64 f16
    off += (size_t)2 * G4 * EMB * sizeof(half_t);
    half_t* WrT    = (half_t*)(ws + off);                            // 2 x 512x128 f16

    prep_weights_kernel<<<(2 * G4 * HID + 255) / 256, 256, 0, stream>>>(
        Wk_f, Wk_b, Wr_f, Wr_b, WkT, WrT);

    lstm_bidir_kernel<<<32, 256, 0, stream>>>(inputs, E, b_f, b_b, WkT, WrT, h_cat);

    logits_kernel<<<(BATCH * TLEN) / 16, 64, 0, stream>>>(h_cat, Wd, bd, logits);
    crf_kernel<<<BATCH, 32, 0, stream>>>(logits, labels, trans, out);
    copy_trans_kernel<<<(TAGS * TAGS + 255) / 256, 256, 0, stream>>>(trans, out);
}
